// MeshEncoder_28269474742812
// MI455X (gfx1250) — matrix operand order repaired and checked by reading, not compiled, not run on hardware
//
#include <hip/hip_runtime.h>
#include <math.h>

// ---------------------------------------------------------------------------
// MeshEncoder GCN for MI455X (gfx1250, wave32, WMMA bf16 16x16x32).
// Pipeline per layer: pack W -> WMMA GEMM (bf16 in, f32 acc) -> zero agg ->
// edge scatter (HW float atomics) -> fused bias+ELU+bf16 epilogue.
// Final layer: full 512-ch segment_sum + ELU + 2-stage max reduce.
// ---------------------------------------------------------------------------

typedef __attribute__((ext_vector_type(16))) __bf16 v16bf;
typedef __attribute__((ext_vector_type(8)))  float  v8f;

#define WGSZ 256

static inline int gdiv(long a, int b) { return (int)((a + (long)b - 1) / b); }

__device__ __forceinline__ float eluf(float x) { return x > 0.f ? x : (expf(x) - 1.f); }

// round-to-nearest-even f32 -> bf16 bits
__device__ __forceinline__ unsigned short f2bf(float f) {
  unsigned u = __float_as_uint(f);
  unsigned r = (u + 0x7FFFu + ((u >> 16) & 1u)) >> 16;
  return (unsigned short)r;
}

__global__ void k_zero(float* __restrict__ p, long n) {
  long i = blockIdx.x * (long)blockDim.x + threadIdx.x;
  if (i < n) p[i] = 0.f;
}

__global__ void k_deg(const int* __restrict__ dst, float* __restrict__ deg, long E) {
  long i = blockIdx.x * (long)blockDim.x + threadIdx.x;
  if (i < E)
    __hip_atomic_fetch_add(&deg[dst[i]], 1.f, __ATOMIC_RELAXED, __HIP_MEMORY_SCOPE_AGENT);
}

__global__ void k_invdeg(const float* __restrict__ deg, float* __restrict__ inv, int n) {
  int i = blockIdx.x * blockDim.x + threadIdx.x;
  if (i < n) inv[i] = 1.f / fmaxf(deg[i], 1.f);
}

// Layer 0: X[N,3] @ W[3,128] done scalar (K=3 too small for WMMA).
__global__ void k_layer0(const float* __restrict__ pos, const float* __restrict__ W,
                         float* __restrict__ S, int n) {
  long i = blockIdx.x * (long)blockDim.x + threadIdx.x;
  int c = (int)(i & 127);
  int node = (int)(i >> 7);
  if (node >= n) return;
  const float* p = pos + (size_t)node * 3;
  S[(size_t)node * 128 + c] = p[0] * W[c] + p[1] * W[128 + c] + p[2] * W[256 + c];
}

// Pack W[Fin,Fout] (f32 row-major) into per-(ktile,coltile) B fragments:
// lane = col(0..15) | khalf(bit4); VGPR v holds K = kt*32 + khalf*8 + 2v + (v>=4 ? 8 : 0), +1.
// Layout: Bp[(((kt*nCol)+ct)*32 + lane)*8 + v], each lane reads 8 consecutive dwords.
__global__ void k_packw(const float* __restrict__ W, unsigned* __restrict__ Bp,
                        int Fin, int Fout, int Kpad, int Fpad) {
  int idx = blockIdx.x * blockDim.x + threadIdx.x;
  int nCol = Fpad >> 4;
  int total = (Kpad >> 5) * nCol * 256;
  if (idx >= total) return;
  int v = idx & 7;
  int lane = (idx >> 3) & 31;
  int tp = idx >> 8;
  int ct = tp % nCol;
  int kt = tp / nCol;
  int col = ct * 16 + (lane & 15);
  int k = kt * 32 + ((lane >> 4) ? 8 : 0) + 2 * v + (v >= 4 ? 8 : 0);
  float lo = (k < Fin && col < Fout) ? W[(size_t)k * Fout + col] : 0.f;
  float hi = (k + 1 < Fin && col < Fout) ? W[(size_t)(k + 1) * Fout + col] : 0.f;
  Bp[idx] = (unsigned)f2bf(lo) | ((unsigned)f2bf(hi) << 16);
}

// WMMA GEMM: S[N,Fpad] = A[N,Kpad](bf16) x Wpacked. 2 waves/block, each wave
// owns one 16x16 output tile; accumulate over Kpad/32 k-steps in f32.
__global__ void __launch_bounds__(64)
k_gemm(const unsigned short* __restrict__ A, const unsigned* __restrict__ Bp,
       float* __restrict__ S, int n, int Kpad, int ldS, int nColTiles) {
  const int lane = threadIdx.x & 31;
  const int wave = threadIdx.x >> 5;
  const int row0 = blockIdx.y * 16;
  const int colTile = blockIdx.x * 2 + wave;
  const int half = lane >> 4;          // lane group (0: K 0..7/16..23, 1: K 8..15/24..31)
  const int lrow = lane & 15;          // A row within tile / D column
  int ar = row0 + lrow; if (ar > n - 1) ar = n - 1;
  const unsigned* arow = (const unsigned*)(A + (size_t)ar * Kpad);
  v8f acc = {0.f, 0.f, 0.f, 0.f, 0.f, 0.f, 0.f, 0.f};
  const int nK = Kpad >> 5;
  for (int kt = 0; kt < nK; ++kt) {
    union { v16bf v; unsigned u[8]; } a, b;
    const int base = kt * 16 + half * 4;           // dwords; kbase elems = half*8
    a.u[0] = arow[base + 0]; a.u[1] = arow[base + 1];
    a.u[2] = arow[base + 2]; a.u[3] = arow[base + 3];
    a.u[4] = arow[base + 8]; a.u[5] = arow[base + 9];
    a.u[6] = arow[base + 10]; a.u[7] = arow[base + 11];
    const unsigned* bq = Bp + (((size_t)kt * nColTiles + colTile) * 32 + lane) * 8;
#pragma unroll
    for (int j = 0; j < 8; ++j) b.u[j] = bq[j];
    acc = __builtin_amdgcn_wmma_f32_16x16x32_bf16(false, a.v, false, b.v,
                                                  (short)0, acc, false, false);
  }
  const int cbase = colTile * 16 + lrow;
#pragma unroll
  for (int r = 0; r < 8; ++r) {
    int row = row0 + r + half * 8;
    if (row < n) S[(size_t)row * ldS + cbase] = acc[r];
  }
}

// Edge scatter: agg[dst, c] += S[src, c] * (useInv ? inv_deg[src] : 1)
__global__ void k_scatter(const float* __restrict__ S, const float* __restrict__ inv,
                          const int* __restrict__ src, const int* __restrict__ dst,
                          float* __restrict__ Agg, long total, int side, int ldS,
                          int useInv) {
  long idx = blockIdx.x * (long)blockDim.x + threadIdx.x;
  if (idx >= total) return;
  int c = (int)(idx % side);
  long e = idx / side;
  int s = src[e], d = dst[e];
  float v = S[(size_t)s * ldS + c];
  if (useInv) v *= inv[s];
  __hip_atomic_fetch_add(&Agg[(size_t)d * side + c], v,
                         __ATOMIC_RELAXED, __HIP_MEMORY_SCOPE_AGENT);
}

// ZERON epilogue: concat(agg, passthrough) + bias -> ELU -> bf16 X (K padded).
__global__ void k_combine(const float* __restrict__ S, const float* __restrict__ Agg,
                          const float* __restrict__ bias, unsigned short* __restrict__ X,
                          int n, int Fout, int ldS, int side, int KpadNext) {
  long idx = blockIdx.x * (long)blockDim.x + threadIdx.x;
  long total = (long)n * KpadNext;
  if (idx >= total) return;
  int c = (int)(idx % KpadNext);
  int node = (int)(idx / KpadNext);
  float y = 0.f;
  if (c < Fout) {
    float v = (c < side ? Agg[(size_t)node * side + c] : S[(size_t)node * ldS + c]) + bias[c];
    y = eluf(v);
  }
  X[idx] = f2bf(y);
}

// Final max over nodes of elu(agg + b): stage 1 partials over node chunks.
__global__ void k_max1(const float* __restrict__ Agg, const float* __restrict__ bias,
                       float* __restrict__ partial, int n, int chunk) {
  int col = blockIdx.x * blockDim.x + threadIdx.x;   // 0..511
  int start = blockIdx.y * chunk;
  int end = start + chunk; if (end > n) end = n;
  float bb = bias[col];
  float m = -2.0f;                                   // elu(x) > -1 always
  for (int i = start; i < end; ++i) {
    float v = eluf(Agg[(size_t)i * 512 + col] + bb);
    m = fmaxf(m, v);
  }
  partial[(size_t)blockIdx.y * 512 + col] = m;
}

__global__ void k_max2(const float* __restrict__ partial, float* __restrict__ out,
                       int nchunks) {
  int col = threadIdx.x;                             // 512 threads
  float m = -2.0f;
  for (int i = 0; i < nchunks; ++i) m = fmaxf(m, partial[(size_t)i * 512 + col]);
  out[col] = m;
}

extern "C" void kernel_launch(void* const* d_in, const int* in_sizes, int n_in,
                              void* d_out, int out_size, void* d_ws, size_t ws_size,
                              hipStream_t stream) {
  (void)n_in; (void)out_size; (void)ws_size;
  const float* positions = (const float*)d_in[0];
  const int*   esrc = (const int*)d_in[1];
  const int*   edst = (const int*)d_in[2];
  const float* W[10]; const float* Bb[10];
  for (int i = 0; i < 10; ++i) { W[i] = (const float*)d_in[3 + i]; Bb[i] = (const float*)d_in[13 + i]; }
  const int  n = in_sizes[0] / 3;
  const long E = in_sizes[1];

  // Workspace carve (256B aligned)
  char* ws = (char*)d_ws;
  size_t off = 0;
  auto carve = [&](size_t bytes) { char* p = ws + off; off = (off + bytes + 255) & ~(size_t)255; return p; };
  float*          deg     = (float*)carve((size_t)n * 4);
  float*          inv     = (float*)carve((size_t)n * 4);
  float*          S       = (float*)carve((size_t)n * 512 * 4);   // support, stride Fpad
  float*          Agg     = (float*)carve((size_t)n * 512 * 4);   // aggregation, stride side
  unsigned short* X       = (unsigned short*)carve((size_t)n * 512 * 2); // bf16 activations
  unsigned*       Wp      = (unsigned*)carve(512 * 1024);         // packed weights
  float*          partial = (float*)carve(256 * 512 * 4);

  const int Fin[10]  = {3, 128, 128, 128, 128, 128, 128, 128, 150, 300};
  const int Fout[10] = {128, 128, 128, 128, 128, 128, 128, 150, 300, 512};
  auto pad32 = [](int x) { return (x + 31) & ~31; };

  // Degrees
  k_zero<<<gdiv(n, WGSZ), WGSZ, 0, stream>>>(deg, n);
  k_deg<<<gdiv(E, WGSZ), WGSZ, 0, stream>>>(edst, deg, E);
  k_invdeg<<<gdiv(n, WGSZ), WGSZ, 0, stream>>>(deg, inv, n);

  for (int i = 0; i <= 9; ++i) {
    const int fi = Fin[i], fo = Fout[i];
    const int Kpad = pad32(fi), Fpad = pad32(fo);
    if (i == 0) {
      long t = (long)n * 128;
      k_layer0<<<gdiv(t, WGSZ), WGSZ, 0, stream>>>(positions, W[0], S, n);
    } else {
      const int nCol = Fpad >> 4;
      long tp = (long)(Kpad >> 5) * nCol * 256;
      k_packw<<<gdiv(tp, WGSZ), WGSZ, 0, stream>>>(W[i], Wp, fi, fo, Kpad, Fpad);
      dim3 g(Fpad / 32, (n + 15) / 16);
      k_gemm<<<g, 64, 0, stream>>>(X, Wp, S, n, Kpad, Fpad, nCol);
    }
    if (i < 9) {
      int side = fo / 3; if (side < 2) side = 2;
      long ta = (long)n * side;
      k_zero<<<gdiv(ta, WGSZ), WGSZ, 0, stream>>>(Agg, ta);
      long tsc = E * (long)side;
      k_scatter<<<gdiv(tsc, WGSZ), WGSZ, 0, stream>>>(S, inv, esrc, edst, Agg,
                                                      tsc, side, Fpad, 1);
      int KpadNext = pad32(fo);
      long tc = (long)n * KpadNext;
      k_combine<<<gdiv(tc, WGSZ), WGSZ, 0, stream>>>(S, Agg, Bb[i], X, n, fo,
                                                     Fpad, side, KpadNext);
    } else {
      long ta = (long)n * 512;
      k_zero<<<gdiv(ta, WGSZ), WGSZ, 0, stream>>>(Agg, ta);
      long tsc = E * 512L;
      k_scatter<<<gdiv(tsc, WGSZ), WGSZ, 0, stream>>>(S, inv, esrc, edst, Agg,
                                                      tsc, 512, 512, 0);
      const int nch = 250;
      const int chunk = (n + nch - 1) / nch;
      dim3 g1(2, nch);
      k_max1<<<g1, 256, 0, stream>>>(Agg, Bb[9], partial, n, chunk);
      k_max2<<<1, 512, 0, stream>>>(partial, (float*)d_out, nch);
    }
  }
}
